// SudokuConstraintAttention_86268713107745
// MI455X (gfx1250) — matrix operand order repaired and checked by reading, not compiled, not run on hardware
//
#include <hip/hip_runtime.h>
#include <hip/hip_bf16.h>

typedef unsigned short u16;
typedef __attribute__((ext_vector_type(16))) __bf16 v16bf;
typedef __attribute__((ext_vector_type(8)))  __bf16 v8bf;
typedef __attribute__((ext_vector_type(8)))  float  v8f;

#define TOKENS (1024 * 81)   // 82944 rows, multiple of 128
#define LDSP   40            // LDS row pitch in bf16 (80B, 16B-aligned, conflict-breaking)

// CDNA5 async global->LDS copy path (ASYNCcnt-tracked), with synchronous fallback.
#if __has_builtin(__builtin_amdgcn_global_load_async_to_lds_b128) && \
    __has_builtin(__builtin_amdgcn_s_wait_asynccnt)
#define USE_ASYNC_COPY 1
#else
#define USE_ASYNC_COPY 0
#endif

#if USE_ASYNC_COPY
// Parameter types per hipcc diagnostic: pointer-to-int4-vector in AS1 (global)
// and the LDS-side analog in AS3.
typedef int gv4i __attribute__((vector_size(16)));
typedef __attribute__((address_space(1))) gv4i* as1_v4p;
typedef __attribute__((address_space(3))) gv4i* as3_v4p;
#endif

__device__ __forceinline__ u16 f2bf(float f) {
  unsigned u = __float_as_uint(f);
  u += 0x7FFFu + ((u >> 16) & 1u);          // round-to-nearest-even
  return (u16)(u >> 16);
}
__device__ __forceinline__ float bf2f(u16 b) {
  return __uint_as_float(((unsigned)b) << 16);
}

// ---------------- fp32 -> bf16 conversion ----------------
__global__ void cvt_f32_bf16(const float* __restrict__ in, u16* __restrict__ out, long long n) {
  long long i = (long long)blockIdx.x * blockDim.x + threadIdx.x;
  long long stride = (long long)gridDim.x * blockDim.x;
  for (; i < n; i += stride) out[i] = f2bf(in[i]);
}

// ---------------- 16B tile-chunk copy: global -> LDS ----------------
__device__ __forceinline__ void copy16(const u16* __restrict__ g, u16* l) {
#if USE_ASYNC_COPY
  __builtin_amdgcn_global_load_async_to_lds_b128((as1_v4p)g, (as3_v4p)l, 0, 0);
#else
  *(uint4*)l = *(const uint4*)g;
#endif
}

// Stage one 128x32 bf16 A-tile and W-tile into LDS (512 chunks, 2/thread each).
__device__ __forceinline__ void copy_tiles(
    const u16* __restrict__ A, const u16* __restrict__ W,
    u16* sA, u16* sB, int bm0, int bn0, int K, int kk, int tid)
{
#pragma unroll
  for (int it = 0; it < 2; ++it) {
    int id  = tid + it * 256;
    int row = id >> 2;
    int c8  = id & 3;
    copy16(A + (size_t)(bm0 + row) * K + kk + c8 * 8, &sA[row * LDSP + c8 * 8]);
    copy16(W + (size_t)(bn0 + row) * K + kk + c8 * 8, &sB[row * LDSP + c8 * 8]);
  }
}

// ---------------- bf16 WMMA GEMM: C[M,N] = A[M,K] * W[N,K]^T + bias ----------------
// Block tile 128x128, 8 waves (4 in M x 2 in N), wave tile 32x64 = 2x4 WMMA tiles.
// Double-buffered LDS, one barrier per K-step: barrier -> issue copy(k+1) -> MMA(k).
__global__ __launch_bounds__(256) void gemm_bf16(
    const u16* __restrict__ A, const u16* __restrict__ W,
    const float* __restrict__ bias, void* __restrict__ outp,
    int K, int ldc, int colOff, int act, int outBf16)
{
  __shared__ __align__(16) u16 sA[2][128 * LDSP];
  __shared__ __align__(16) u16 sB[2][128 * LDSP];

  const int tid  = threadIdx.x;
  const int lane = tid & 31;
  const int wave = tid >> 5;
  const int wm   = wave >> 1;            // 0..3 -> M offset wm*32
  const int wn   = wave & 1;             // 0..1 -> N offset wn*64
  const int bm0  = blockIdx.y * 128;
  const int bn0  = blockIdx.x * 128;
  const int lr   = lane & 15;
  const int hi   = lane >> 4;

  const v8f vzero = {0.f, 0.f, 0.f, 0.f, 0.f, 0.f, 0.f, 0.f};
  v8f acc[2][4];
#pragma unroll
  for (int i = 0; i < 2; ++i)
#pragma unroll
    for (int j = 0; j < 4; ++j) acc[i][j] = vzero;

  const int nk = K >> 5;
  copy_tiles(A, W, sA[0], sB[0], bm0, bn0, K, 0, tid);

  for (int t = 0; t < nk; ++t) {
#if USE_ASYNC_COPY
    __builtin_amdgcn_s_wait_asynccnt(0);   // tile t fully landed in LDS
#endif
    __syncthreads();
    if (t + 1 < nk)
      copy_tiles(A, W, sA[(t + 1) & 1], sB[(t + 1) & 1], bm0, bn0, K, (t + 1) << 5, tid);

    const u16* cA = sA[t & 1];
    const u16* cB = sB[t & 1];

    v16bf af[2], bfr[4];
#pragma unroll
    for (int i = 0; i < 2; ++i) {
      // A-matrix 16-bit layout: hi=0 -> K 0..7 & 16..23, hi=1 -> K 8..15 & 24..31
      const u16* p = &cA[(wm * 32 + i * 16 + lr) * LDSP];
      v8bf lo = *(const v8bf*)(p + hi * 8);
      v8bf hc = *(const v8bf*)(p + 16 + hi * 8);
      af[i] = __builtin_shufflevector(lo, hc, 0,1,2,3,4,5,6,7,8,9,10,11,12,13,14,15);
    }
#pragma unroll
    for (int j = 0; j < 4; ++j) {
      // B-matrix 16-bit layout: hi=0 -> K 0..7 & 16..23, hi=1 -> K 8..15 & 24..31
      const u16* p = &cB[(wn * 64 + j * 16 + lr) * LDSP + hi * 8];
      v8bf lo = *(const v8bf*)(p);
      v8bf hc = *(const v8bf*)(p + 16);
      bfr[j] = __builtin_shufflevector(lo, hc, 0,1,2,3,4,5,6,7,8,9,10,11,12,13,14,15);
    }
#pragma unroll
    for (int i = 0; i < 2; ++i)
#pragma unroll
      for (int j = 0; j < 4; ++j)
        acc[i][j] = __builtin_amdgcn_wmma_f32_16x16x32_bf16(
            false, af[i], false, bfr[j], (short)0, acc[i][j], false, false);
  }

  // Epilogue: C/D layout -> lane = column (lr), VGPR e = row hi*8+e
  const int mh = hi * 8;
#pragma unroll
  for (int i = 0; i < 2; ++i) {
#pragma unroll
    for (int j = 0; j < 4; ++j) {
      int gn = bn0 + wn * 64 + j * 16 + lr;
      float bb = bias[gn];
      size_t col = (size_t)colOff + gn;
#pragma unroll
      for (int e = 0; e < 8; ++e) {
        int gm = bm0 + wm * 32 + i * 16 + mh + e;
        float v = acc[i][j][e] + bb;
        if (act == 1) v = 0.5f * v * (1.0f + erff(v * 0.70710678118654752f));  // exact GELU
        if (outBf16) ((u16*)outp)[(size_t)gm * ldc + col] = f2bf(v);
        else         ((float*)outp)[(size_t)gm * ldc + col] = v;
      }
    }
  }
}

// ---------------- 9-token attention (row/col/box via index gather) ----------------
__device__ __forceinline__ int tok_idx(int mode, int g, int s) {
  if (mode == 0) return g * 9 + s;                      // row
  if (mode == 1) return s * 9 + g;                      // col
  int r = (g / 3) * 3 + (s / 3);                        // box
  int c = (g % 3) * 3 + (s % 3);
  return r * 9 + c;
}

__global__ __launch_bounds__(256) void attn9(
    const u16* __restrict__ qkv, u16* __restrict__ attout, int mode)
{
  __shared__ float sQ[8][9][64];
  __shared__ float sK[8][9][64];
  __shared__ float sV[8][9][64];
  __shared__ float sS[8][9][9];

  const int bg = blockIdx.x;            // b*9 + g
  const int b = bg / 9, g = bg % 9;
  const int h = threadIdx.x >> 5;       // one wave per head
  const int lane = threadIdx.x & 31;

  int tok[9];
#pragma unroll
  for (int s = 0; s < 9; ++s) tok[s] = b * 81 + tok_idx(mode, g, s);

#pragma unroll
  for (int s = 0; s < 9; ++s) {
    const u16* p = qkv + (size_t)tok[s] * 1536 + h * 64;
    sQ[h][s][lane]      = bf2f(p[lane]);
    sQ[h][s][lane + 32] = bf2f(p[lane + 32]);
    sK[h][s][lane]      = bf2f(p[512 + lane]);
    sK[h][s][lane + 32] = bf2f(p[512 + lane + 32]);
    sV[h][s][lane]      = bf2f(p[1024 + lane]);
    sV[h][s][lane + 32] = bf2f(p[1024 + lane + 32]);
  }
  __syncthreads();

  for (int p = lane; p < 81; p += 32) {
    int q = p / 9, k = p % 9;
    float a = 0.f;
#pragma unroll
    for (int d = 0; d < 64; ++d) a += sQ[h][q][d] * sK[h][k][d];
    sS[h][q][k] = a * 0.125f;           // 1/sqrt(64)
  }
  __syncthreads();

  if (lane < 9) {
    float m = sS[h][lane][0];
#pragma unroll
    for (int k = 1; k < 9; ++k) m = fmaxf(m, sS[h][lane][k]);
    float e[9], sum = 0.f;
#pragma unroll
    for (int k = 0; k < 9; ++k) { e[k] = expf(sS[h][lane][k] - m); sum += e[k]; }
    float inv = 1.0f / sum;
#pragma unroll
    for (int k = 0; k < 9; ++k) sS[h][lane][k] = e[k] * inv;
  }
  __syncthreads();

#pragma unroll
  for (int s = 0; s < 9; ++s) {
    float o0 = 0.f, o1 = 0.f;
#pragma unroll
    for (int k = 0; k < 9; ++k) {
      float pw = sS[h][s][k];
      o0 += pw * sV[h][k][lane];
      o1 += pw * sV[h][k][lane + 32];
    }
    u16* op = attout + (size_t)tok[s] * 512 + h * 64;
    op[lane]      = f2bf(o0);
    op[lane + 32] = f2bf(o1);
  }
}

// ---------------- residual + LayerNorm (fp32) ----------------
__global__ __launch_bounds__(256) void ln512(
    const float* __restrict__ x, const float* __restrict__ mixed,
    const float* __restrict__ gamma, const float* __restrict__ beta,
    float* __restrict__ out)
{
  const int t = blockIdx.x;
  const int tid = threadIdx.x;
  __shared__ float s1[8], s2[8];

  const float* xr = x + (size_t)t * 512;
  const float* mr = mixed + (size_t)t * 512;
  float y0 = xr[tid] + mr[tid];
  float y1 = xr[tid + 256] + mr[tid + 256];
  float s = y0 + y1, ss = y0 * y0 + y1 * y1;
  for (int o = 16; o > 0; o >>= 1) { s += __shfl_down(s, o); ss += __shfl_down(ss, o); }
  int wave = tid >> 5, lane = tid & 31;
  if (lane == 0) { s1[wave] = s; s2[wave] = ss; }
  __syncthreads();
  if (tid == 0) {
    float a = 0.f, bsum = 0.f;
    for (int i = 0; i < 8; ++i) { a += s1[i]; bsum += s2[i]; }
    s1[0] = a; s2[0] = bsum;
  }
  __syncthreads();
  float mean = s1[0] * (1.f / 512.f);
  float var  = s2[0] * (1.f / 512.f) - mean * mean;
  float inv  = rsqrtf(var + 1e-5f);
  out[(size_t)t * 512 + tid]       = (y0 - mean) * inv * gamma[tid] + beta[tid];
  out[(size_t)t * 512 + tid + 256] = (y1 - mean) * inv * gamma[tid + 256] + beta[tid + 256];
}

// ---------------- launcher ----------------
extern "C" void kernel_launch(void* const* d_in, const int* in_sizes, int n_in,
                              void* d_out, int out_size, void* d_ws, size_t ws_size,
                              hipStream_t stream)
{
  (void)in_sizes; (void)n_in; (void)out_size; (void)ws_size;

  const float* x = (const float*)d_in[0];
  const float* wqkv[3] = {(const float*)d_in[1], (const float*)d_in[5], (const float*)d_in[9]};
  const float* bqkv[3] = {(const float*)d_in[2], (const float*)d_in[6], (const float*)d_in[10]};
  const float* wo[3]   = {(const float*)d_in[3], (const float*)d_in[7], (const float*)d_in[11]};
  const float* bo[3]   = {(const float*)d_in[4], (const float*)d_in[8], (const float*)d_in[12]};
  const float* w1 = (const float*)d_in[13];
  const float* b1 = (const float*)d_in[14];
  const float* w2 = (const float*)d_in[15];
  const float* b2 = (const float*)d_in[16];
  const float* gamma = (const float*)d_in[17];
  const float* beta  = (const float*)d_in[18];

  const size_t M = TOKENS;
  // Workspace layout (u16 units):
  //   xb [M*512] | atto [M*512] | comb [M*1536] | qkvb [M*1536] | weights
  // Liveness-based reuse: hbuf overlays qkvb after attentions; mixed (f32)
  // overlays xb+atto (exactly M*2048 bytes) after QKV GEMMs are done.
  u16* xb   = (u16*)d_ws;
  u16* atto = xb + M * 512;
  u16* comb = atto + M * 512;
  u16* qkvb = comb + M * 1536;
  u16* wp   = qkvb + M * 1536;
  u16* wq[3], *wob[3];
  for (int p = 0; p < 3; ++p) { wq[p]  = wp; wp += 1536 * 512; }
  for (int p = 0; p < 3; ++p) { wob[p] = wp; wp += 512 * 512; }
  u16* w1b = wp; wp += 1024 * 1536;
  u16* w2b = wp; wp += 512 * 1024;
  u16* hbuf = qkvb;            // MLP hidden (M*1024 <= M*1536)
  float* mixed = (float*)d_ws; // M*512 f32 over xb+atto

  dim3 blk(256);
  auto cgrid = [](long long n) {
    long long g = (n + 255) / 256;
    if (g > 65535) g = 65535;
    return dim3((unsigned)g);
  };

  // bf16 conversions
  cvt_f32_bf16<<<cgrid((long long)M * 512), blk, 0, stream>>>(x, xb, (long long)M * 512);
  for (int p = 0; p < 3; ++p) {
    cvt_f32_bf16<<<cgrid(1536 * 512), blk, 0, stream>>>(wqkv[p], wq[p], 1536 * 512);
    cvt_f32_bf16<<<cgrid(512 * 512),  blk, 0, stream>>>(wo[p], wob[p], 512 * 512);
  }
  cvt_f32_bf16<<<cgrid(1024 * 1536), blk, 0, stream>>>(w1, w1b, 1024 * 1536);
  cvt_f32_bf16<<<cgrid(512 * 1024),  blk, 0, stream>>>(w2, w2b, 512 * 1024);

  const unsigned MB = (unsigned)(M / 128);   // 648
  for (int p = 0; p < 3; ++p) {
    // QKV projection: [M,512] x [1536,512]^T -> qkv bf16 [M,1536]
    gemm_bf16<<<dim3(1536 / 128, MB), blk, 0, stream>>>(xb, wq[p], bqkv[p], qkvb,
                                                        512, 1536, 0, 0, 1);
    // 9-token attention with row/col/box gather
    attn9<<<dim3(1024 * 9), blk, 0, stream>>>(qkvb, atto, p);
    // Output projection, written into the concat buffer at column offset p*512
    gemm_bf16<<<dim3(512 / 128, MB), blk, 0, stream>>>(atto, wob[p], bo[p], comb,
                                                       512, 1536, p * 512, 0, 1);
  }
  // MLP: gelu(comb @ w1^T + b1) -> hbuf ; hbuf @ w2^T + b2 -> mixed (f32)
  gemm_bf16<<<dim3(1024 / 128, MB), blk, 0, stream>>>(comb, w1b, b1, hbuf,
                                                      1536, 1024, 0, 1, 1);
  gemm_bf16<<<dim3(512 / 128, MB), blk, 0, stream>>>(hbuf, w2b, b2, mixed,
                                                     1024, 512, 0, 0, 0);
  // Residual + LayerNorm -> d_out (fp32)
  ln512<<<dim3((unsigned)M), blk, 0, stream>>>(x, mixed, gamma, beta, (float*)d_out);
}
